// AbstractAttentionLayer_73048803770796
// MI455X (gfx1250) — matrix operand order repaired and checked by reading, not compiled, not run on hardware
//
#include <hip/hip_runtime.h>
#include <cstdint>
#include <cstddef>

// ---------------------------------------------------------------------------
// Causal attention (B=8, S=2048, D=U=1024) for MI455X / gfx1250 (wave32).
// All heavy math runs through v_wmma_f32_16x16x32_f16 (fp32 accumulate).
// metric is folded into Wq:  Q' = X1 @ (Wq@metric) + bq@metric.
// Register-blocked 2x2: each wave computes a 32x32 output block, reusing each
// A fragment across 2 N-tiles and each B fragment across 2 M-tiles.
// ---------------------------------------------------------------------------

typedef _Float16 half_t;
typedef __attribute__((ext_vector_type(16))) _Float16 v16h;
typedef __attribute__((ext_vector_type(8)))  _Float16 v8h;
typedef __attribute__((ext_vector_type(8)))  float    v8f;

#define BN 8
#define SN 2048
#define DN 1024
#define UN 1024

__device__ __forceinline__ v16h cat8(v8h lo, v8h hi) {
  return __builtin_shufflevector(lo, hi, 0,1,2,3,4,5,6,7,8,9,10,11,12,13,14,15);
}

// A operand (16x32 f16) from a row-major [M,K] source.
// Layout used here (contiguous k, same as load_Bt): lane L holds row m=L&15; half h=L>>4; element e ->
//   K = k0 + h*16 + e; i.e. one contiguous 16-half chunk.
__device__ __forceinline__ v16h load_A(const half_t* A, int lda, int m0, int k0, int lane) {
  int m = lane & 15, h = lane >> 4;
  const half_t* p = A + (size_t)(m0 + m) * lda + k0 + h * 16;
  v8h lo = *(const v8h*)(p);        // K = k0+h*16   .. +7
  v8h hi = *(const v8h*)(p + 8);    // K = k0+h*16+8 .. +7
  return cat8(lo, hi);
}

// B operand (32x16 f16). Source "Bt" stores each output column's K-row
// contiguously: Bt[n][k], leading dim ldb.
// Layout used here (contiguous k, same as load_A): lane L holds column n=L&15; element e -> K = k0 + (L>>4)*16 + e.
__device__ __forceinline__ v16h load_Bt(const half_t* Bt, int ldb, int n0, int k0, int lane) {
  int n = lane & 15, h = lane >> 4;
  const half_t* p = Bt + (size_t)(n0 + n) * ldb + k0 + h * 16;
  v8h lo = *(const v8h*)(p);
  v8h hi = *(const v8h*)(p + 8);
  return cat8(lo, hi);
}

__device__ __forceinline__ v8f wmma16(v16h a, v16h b, v8f c) {
  // (neg_a, A, neg_b, B, c_mod, C, reuse_a, reuse_b)
  return __builtin_amdgcn_wmma_f32_16x16x32_f16(false, a, false, b, (short)0, c, false, false);
}

// ---------------------------------------------------------------------------
// Prep kernels
// ---------------------------------------------------------------------------

__global__ void k_f32_to_f16(const float* __restrict__ in, half_t* __restrict__ out, size_t n) {
  size_t i = (size_t)blockIdx.x * blockDim.x + threadIdx.x;
  size_t stride = (size_t)gridDim.x * blockDim.x;
  for (; i < n; i += stride) out[i] = (half_t)in[i];
}

// out[c][r] = (f16) in[r][c];  in: R x C fp32 row-major, out: C x R f16 row-major.
__global__ void k_transpose_f16(const float* __restrict__ in, half_t* __restrict__ out,
                                int R, int C) {
  size_t i = (size_t)blockIdx.x * blockDim.x + threadIdx.x;
  size_t n = (size_t)R * C;
  size_t stride = (size_t)gridDim.x * blockDim.x;
  for (; i < n; i += stride) {
    int c = (int)(i / R);
    int r = (int)(i % R);
    out[i] = (half_t)in[(size_t)r * C + c];   // coalesced writes
  }
}

// bq_eff[v] = sum_u bq[u] * metric[u][v]   (1M MACs, negligible)
__global__ void k_bias_metric(const float* __restrict__ bq, const float* __restrict__ metric,
                              float* __restrict__ bqe) {
  int v = blockIdx.x * blockDim.x + threadIdx.x;
  if (v < UN) {
    float s = 0.f;
    for (int u = 0; u < UN; ++u) s += bq[u] * metric[(size_t)u * UN + v];
    bqe[v] = s;
  }
}

// ---------------------------------------------------------------------------
// Generic f16 WMMA GEMM: C[M,N](f16) = A[M,K] @ Bt^T + bias  (Bt is [N,K]).
// Block = 256 threads = 8 waves arranged 2(M) x 4(N); each wave owns a 32x32
// C block (2x2 WMMA tiles).  grid = (N/128, M/64).
// tS>0: store batch-transposed  C[(row/tS)*N + n][row%tS]  (used for V^T).
// ---------------------------------------------------------------------------
__global__ void k_gemm_f16(const half_t* __restrict__ A, const half_t* __restrict__ Bt,
                           const float* __restrict__ bias, half_t* __restrict__ C,
                           int M, int N, int K, int tS) {
  int lane = threadIdx.x & 31;
  int wave = threadIdx.x >> 5;
  int wm = wave >> 2;                       // 0..1
  int wn = wave & 3;                        // 0..3
  int m0 = blockIdx.y * 64 + wm * 32;
  int n0 = blockIdx.x * 128 + wn * 32;

  v8f acc[2][2] = {};
#pragma unroll 2
  for (int k0 = 0; k0 < K; k0 += 32) {
    v16h a0 = load_A(A, K, m0,      k0, lane);
    v16h a1 = load_A(A, K, m0 + 16, k0, lane);
    v16h b0 = load_Bt(Bt, K, n0,      k0, lane);
    v16h b1 = load_Bt(Bt, K, n0 + 16, k0, lane);
    acc[0][0] = wmma16(a0, b0, acc[0][0]);
    acc[0][1] = wmma16(a0, b1, acc[0][1]);
    acc[1][0] = wmma16(a1, b0, acc[1][0]);
    acc[1][1] = wmma16(a1, b1, acc[1][1]);
  }

  int h = lane >> 4;
  int nl = lane & 15;
  float bb[2];
  bb[0] = bias ? bias[n0 + nl]      : 0.0f;
  bb[1] = bias ? bias[n0 + 16 + nl] : 0.0f;
#pragma unroll
  for (int mt = 0; mt < 2; ++mt) {
#pragma unroll
    for (int nt = 0; nt < 2; ++nt) {
      int n = n0 + nt * 16 + nl;
#pragma unroll
      for (int r = 0; r < 8; ++r) {
        int row = m0 + mt * 16 + r + h * 8;      // C/D layout: M = vgpr + 8*half
        float v = acc[mt][nt][r] + bb[nt];
        if (tS == 0) {
          C[(size_t)row * N + n] = (half_t)v;
        } else {                                  // per-batch transposed store
          int b = row / tS, j = row % tS;
          C[((size_t)b * N + n) * tS + j] = (half_t)v;
        }
      }
    }
  }
}

// ---------------------------------------------------------------------------
// Scores + causal mask + single-pass exp + row sums.
// One wave per 32-row strip (2 M-tiles), j processed 32 at a time (2 N-tiles).
// With 32-aligned strips the causal loop bound (j0 <= i0+16, step 32) ends at
// j0 == i0, so P is written for exactly j in [0, i0+32) -- no pad tile needed.
// P[b][i][j] = exp(q.k/32) (f16), masked entries -> 0.  Linv[b][i] = 1/rowsum.
// grid = (S/32/8, B), block = 256.
// ---------------------------------------------------------------------------
__global__ void k_scores(const half_t* __restrict__ Q, const half_t* __restrict__ Kh,
                         half_t* __restrict__ P, float* __restrict__ Linv) {
  const float scale = 0.03125f;   // 1/sqrt(U) = 1/32
  int lane = threadIdx.x & 31;
  int wave = threadIdx.x >> 5;
  int t  = blockIdx.x * 8 + wave;       // 32-row strip index
  int b  = blockIdx.y;
  int i0 = t * 32;
  const half_t* Qb = Q + (size_t)b * SN * UN;
  const half_t* Kb = Kh + (size_t)b * SN * UN;
  half_t* Pb = P + (size_t)b * SN * SN;

  int h = lane >> 4;
  int j_lane = lane & 15;
  float lsum[2][8] = {};

  for (int j0 = 0; j0 <= i0 + 16; j0 += 32) {   // uniform per wave -> EXEC all-ones
    v8f acc[2][2] = {};
#pragma unroll 2
    for (int k0 = 0; k0 < UN; k0 += 32) {
      v16h a0 = load_A(Qb, UN, i0,      k0, lane);
      v16h a1 = load_A(Qb, UN, i0 + 16, k0, lane);
      v16h b0 = load_Bt(Kb, UN, j0,      k0, lane);  // K rows = B columns of Q@K^T
      v16h b1 = load_Bt(Kb, UN, j0 + 16, k0, lane);
      acc[0][0] = wmma16(a0, b0, acc[0][0]);
      acc[0][1] = wmma16(a0, b1, acc[0][1]);
      acc[1][0] = wmma16(a1, b0, acc[1][0]);
      acc[1][1] = wmma16(a1, b1, acc[1][1]);
    }
#pragma unroll
    for (int mt = 0; mt < 2; ++mt) {
#pragma unroll
      for (int nt = 0; nt < 2; ++nt) {
#pragma unroll
        for (int r = 0; r < 8; ++r) {
          int i = i0 + mt * 16 + r + h * 8;
          int j = j0 + nt * 16 + j_lane;
          float s = acc[mt][nt][r] * scale;
          float e = (j <= i) ? __expf(s) : 0.0f;  // causal mask == exp(-1e9) == 0
          lsum[mt][r] += e;
          Pb[(size_t)i * SN + j] = (half_t)e;
        }
      }
    }
  }

  // Row sums: butterfly across the 16 lanes of each half-wave.
#pragma unroll
  for (int mt = 0; mt < 2; ++mt) {
#pragma unroll
    for (int r = 0; r < 8; ++r) {
      float v = lsum[mt][r];
      v += __shfl_xor(v, 1, 32);
      v += __shfl_xor(v, 2, 32);
      v += __shfl_xor(v, 4, 32);
      v += __shfl_xor(v, 8, 32);
      if ((lane & 15) == 0)
        Linv[(size_t)b * SN + i0 + mt * 16 + r + h * 8] = 1.0f / v;
    }
  }
}

// ---------------------------------------------------------------------------
// Out[b][i][u] = (P[b][i][:] @ V[b][:][u]) * Linv[b][i]
// Vt is [B][U][S] so B-operand rows are contiguous.  Each wave: 32x32 block.
// Causal: P strip [i0,i0+32) is valid for j in [0, i0+32) -> steps = i0/32+1.
// grid = (U/256, S/32, B), block = 256 (8 waves along U).
// ---------------------------------------------------------------------------
__global__ void k_pv(const half_t* __restrict__ P, const half_t* __restrict__ Vt,
                     const float* __restrict__ Linv, float* __restrict__ Out) {
  int lane = threadIdx.x & 31;
  int wave = threadIdx.x >> 5;
  int u0 = (blockIdx.x * 8 + wave) * 32;
  int i0 = blockIdx.y * 32;
  int b  = blockIdx.z;
  const half_t* Pb = P  + (size_t)b * SN * SN;
  const half_t* Vb = Vt + (size_t)b * UN * SN;

  int steps = (i0 >> 5) + 1;           // K=32 chunks of the causal j-range
  v8f acc[2][2] = {};
#pragma unroll 2
  for (int p = 0; p < steps; ++p) {
    int j0 = p * 32;
    v16h a0 = load_A(Pb, SN, i0,      j0, lane);
    v16h a1 = load_A(Pb, SN, i0 + 16, j0, lane);
    v16h b0 = load_Bt(Vb, SN, u0,      j0, lane);
    v16h b1 = load_Bt(Vb, SN, u0 + 16, j0, lane);
    acc[0][0] = wmma16(a0, b0, acc[0][0]);
    acc[0][1] = wmma16(a0, b1, acc[0][1]);
    acc[1][0] = wmma16(a1, b0, acc[1][0]);
    acc[1][1] = wmma16(a1, b1, acc[1][1]);
  }

  int h = lane >> 4;
  int nl = lane & 15;
#pragma unroll
  for (int mt = 0; mt < 2; ++mt) {
#pragma unroll
    for (int r = 0; r < 8; ++r) {
      int i = i0 + mt * 16 + r + h * 8;
      float li = Linv[(size_t)b * SN + i];
#pragma unroll
      for (int nt = 0; nt < 2; ++nt) {
        int u = u0 + nt * 16 + nl;
        Out[((size_t)b * SN + i) * UN + u] = acc[mt][nt][r] * li;
      }
    }
  }
}

// ---------------------------------------------------------------------------
// Host launcher
// ---------------------------------------------------------------------------
extern "C" void kernel_launch(void* const* d_in, const int* in_sizes, int n_in,
                              void* d_out, int out_size, void* d_ws, size_t ws_size,
                              hipStream_t stream) {
  (void)in_sizes; (void)n_in; (void)out_size; (void)ws_size;
  const float* X1     = (const float*)d_in[0];
  const float* X2     = (const float*)d_in[1];
  const float* Wq     = (const float*)d_in[2];
  const float* bq     = (const float*)d_in[3];
  const float* Wk     = (const float*)d_in[4];
  const float* bk     = (const float*)d_in[5];
  const float* Wv     = (const float*)d_in[6];
  const float* bv     = (const float*)d_in[7];
  const float* metric = (const float*)d_in[8];
  float* Out = (float*)d_out;

  const size_t nX  = (size_t)BN * SN * DN;   // 16,777,216
  const size_t nW  = (size_t)DN * UN;        //  1,048,576
  const size_t nP  = (size_t)BN * SN * SN;   // 33,554,432

  // Workspace carve (~245 MB, 256B aligned).
  char* w = (char*)d_ws;
  auto carve = [&](size_t bytes) -> void* {
    void* p = (void*)w;
    w += (bytes + 255) & ~(size_t)255;
    return p;
  };
  half_t* X1h   = (half_t*)carve(2 * nX);
  half_t* X2h   = (half_t*)carve(2 * nX);
  half_t* Wq16  = (half_t*)carve(2 * nW);   // Wq row-major f16
  half_t* Mt    = (half_t*)carve(2 * nW);   // metric^T f16
  half_t* Wkt   = (half_t*)carve(2 * nW);   // Wk^T f16
  half_t* Wvt   = (half_t*)carve(2 * nW);   // Wv^T f16
  half_t* Wqet  = (half_t*)carve(2 * nW);   // (Wq@metric)^T f16
  float*  bqe   = (float*)carve(4 * UN);
  half_t* Qh    = (half_t*)carve(2 * (size_t)BN * SN * UN);
  half_t* Kh    = (half_t*)carve(2 * (size_t)BN * SN * UN);
  half_t* Vth   = (half_t*)carve(2 * (size_t)BN * UN * SN);  // [B][U][S]
  half_t* Pbuf  = (half_t*)carve(2 * nP);
  float*  Linv  = (float*)carve(4 * (size_t)BN * SN);

  // 1) fp32 -> f16 converts + weight transposes
  k_f32_to_f16<<<dim3(4096), dim3(256), 0, stream>>>(X1, X1h, nX);
  k_f32_to_f16<<<dim3(4096), dim3(256), 0, stream>>>(X2, X2h, nX);
  k_f32_to_f16<<<dim3(1024), dim3(256), 0, stream>>>(Wq, Wq16, nW);
  k_transpose_f16<<<dim3(1024), dim3(256), 0, stream>>>(metric, Mt, UN, UN);
  k_transpose_f16<<<dim3(1024), dim3(256), 0, stream>>>(Wk, Wkt, DN, UN);
  k_transpose_f16<<<dim3(1024), dim3(256), 0, stream>>>(Wv, Wvt, DN, UN);

  // 2) Wq_eff^T[v][d] = sum_u metric[u][v] * Wq[d][u]  == Mt @ (Wq16 as Bt)
  k_gemm_f16<<<dim3(DN / 128, UN / 64), dim3(256), 0, stream>>>(
      Mt, Wq16, (const float*)nullptr, Wqet, UN, DN, UN, 0);
  k_bias_metric<<<dim3(UN / 256), dim3(256), 0, stream>>>(bq, metric, bqe);

  // 3) Projections: Q = X1@Wq_eff + bq_eff; K = X2@Wk + bk; V^T from X2@Wv + bv
  k_gemm_f16<<<dim3(UN / 128, (BN * SN) / 64), dim3(256), 0, stream>>>(
      X1h, Wqet, bqe, Qh, BN * SN, UN, DN, 0);
  k_gemm_f16<<<dim3(UN / 128, (BN * SN) / 64), dim3(256), 0, stream>>>(
      X2h, Wkt, bk, Kh, BN * SN, UN, DN, 0);
  k_gemm_f16<<<dim3(UN / 128, (BN * SN) / 64), dim3(256), 0, stream>>>(
      X2h, Wvt, bv, Vth, BN * SN, UN, DN, SN);   // batch-transposed store

  // 4) Causal scores -> P (f16) + 1/rowsum
  k_scores<<<dim3(SN / 32 / 8, BN), dim3(256), 0, stream>>>(Qh, Kh, Pbuf, Linv);

  // 5) Out = (P @ V) * Linv
  k_pv<<<dim3(UN / 256, SN / 32, BN), dim3(256), 0, stream>>>(Pbuf, Vth, Linv, Out);
}
